// AttentionBlock_89713276879184
// MI455X (gfx1250) — matrix operand order repaired and checked by reading, not compiled, not run on hardware
//
#include <hip/hip_runtime.h>
#include <hip/hip_bf16.h>

// ---------------------------------------------------------------------------
// MI455X (gfx1250) implementation of the two-stage graph-attention layer.
//
//  * All edge/node MLP GEMMs run through V_WMMA_F32_16X16X32_F16 (f16 in,
//    fp32 accumulate).  K padded 340 -> 352.  Weights are pre-transposed to
//    Wt[128][Kp] f16 so each lane's B fragment is two contiguous 16B global
//    loads (L2-resident).  A tiles (64 rows x 32 K) are staged through LDS
//    with one global_load_b128 + ds_store_b128 per thread per K-step
//    (40-half row stride -> 16B aligned, conflict-free b128 reads).
//    Each wave holds 4 accumulators (64x16 tile) and reuses the B fragment
//    across 4 v_wmma per K-step.
//  * LayerNorm+ReLU, sigmoid edge weights, segment softmax, and scatter
//    aggregation stay fp32 (monotone-uint atomicMax for segment max,
//    global_atomic_add_f32 for sums).
//
// Input flattening assumption (dict insertion order, nested dicts in
// insertion order, each _mlp_params as w1,b1,g,b,w2,b2):
//   0:h 1:x 2:edge_attr 3:mask_ligand
//   x2h.hk: 4..9   x2h.hv: 10..15   x2h.hq: 16..21
//   22:x2h.ew_w 23:x2h.ew_b        x2h.out: 24..29
//   h2x.xk: 30..35  h2x.xv: 36..41 h2x.xq: 42..47
//   48:h2x.ew_w 49:h2x.ew_b
//   50:edge_index (int32 [2,E])
// Output: [N*128] x2h_out followed by [N*3] x_new (f32).
// ---------------------------------------------------------------------------

typedef __attribute__((ext_vector_type(16))) _Float16 v16h;
typedef __attribute__((ext_vector_type(8)))  _Float16 v8h;
typedef __attribute__((ext_vector_type(8)))  float    v8f;

#define HID 128
#define HEADS 4
#define HEAD_DIM 32
#define EF 4
#define NOFF 20
#define KV_RAW 340
#define KV_PAD 352
#define INV_SQRT_D 0.17677669529663687f
#define SA_STRIDE 40   // halfs; 80B row stride: b128-aligned & conflict-free

static __host__ __device__ inline long long cdivll(long long a, long long b) { return (a + b - 1) / b; }

// ---------------------------------------------------------------------------
__global__ void zero_f32(float* __restrict__ p, long long n) {
  long long i = (long long)blockIdx.x * blockDim.x + threadIdx.x;
  if (i < n) p[i] = 0.0f;
}

__global__ void f32_to_f16(const float* __restrict__ in, _Float16* __restrict__ out, long long n) {
  long long i = (long long)blockIdx.x * blockDim.x + threadIdx.x;
  if (i < n) out[i] = (_Float16)in[i];
}

// Transpose/convert a [Kin,128] fp32 weight into Wt[128,Kp] f16 (pad K with 0).
__global__ void transpose_w_f16(const float* __restrict__ w, _Float16* __restrict__ wt,
                                int Kin, int Kp) {
  long long i = (long long)blockIdx.x * blockDim.x + threadIdx.x;
  long long total = (long long)HID * Kp;
  if (i >= total) return;
  int n = (int)(i / Kp);
  int k = (int)(i - (long long)n * Kp);
  wt[i] = (k < Kin) ? (_Float16)w[(long long)k * HID + n] : (_Float16)0.0f;
}

// ---------------------------------------------------------------------------
// Per-edge geometry: rel_x, gaussian smearing -> r_feat (f16, [E,80]),
// plus both sigmoid edge-weight scalars.
// ---------------------------------------------------------------------------
__global__ void edge_geom(const float* __restrict__ x, const float* __restrict__ ea,
                          const int* __restrict__ ei,
                          const float* __restrict__ eww1, const float* __restrict__ ewb1,
                          const float* __restrict__ eww2, const float* __restrict__ ewb2,
                          _Float16* __restrict__ rfeat16, float* __restrict__ relx,
                          float* __restrict__ ew1, float* __restrict__ ew2, int E) {
  int e = blockIdx.x * blockDim.x + threadIdx.x;
  if (e >= E) return;
  const float offs[NOFF] = {0.f, 1.f, 1.25f, 1.5f, 1.75f, 2.f, 2.25f, 2.5f, 2.75f, 3.f,
                            3.5f, 4.f, 4.5f, 5.f, 5.5f, 6.f, 7.f, 8.f, 9.f, 10.f};
  int s = ei[e], d = ei[E + e];
  float rx = x[d * 3 + 0] - x[s * 3 + 0];
  float ry = x[d * 3 + 1] - x[s * 3 + 1];
  float rz = x[d * 3 + 2] - x[s * 3 + 2];
  relx[e * 3 + 0] = rx; relx[e * 3 + 1] = ry; relx[e * 3 + 2] = rz;
  float dist = sqrtf(rx * rx + ry * ry + rz * rz);
  float a[EF];
#pragma unroll
  for (int f = 0; f < EF; ++f) a[f] = ea[e * EF + f];
  float acc1 = 0.f, acc2 = 0.f;
#pragma unroll
  for (int j = 0; j < NOFF; ++j) {
    float dd = dist - offs[j];
    float g = expf(-0.5f * dd * dd);
#pragma unroll
    for (int f = 0; f < EF; ++f) {
      float rf = a[f] * g;
      int idx = f * NOFF + j;
      rfeat16[(long long)e * (EF * NOFF) + idx] = (_Float16)rf;
      acc1 += rf * eww1[idx];
      acc2 += rf * eww2[idx];
    }
  }
  ew1[e] = 1.0f / (1.0f + expf(-(acc1 + ewb1[0])));
  ew2[e] = 1.0f / (1.0f + expf(-(acc2 + ewb2[0])));
}

// ---------------------------------------------------------------------------
// kv = [edge_attr(4), r_feat(80), h16[dst](128), h16[src](128), pad(12)]
// ---------------------------------------------------------------------------
__global__ void build_kv(const float* __restrict__ ea, const _Float16* __restrict__ rfeat16,
                         const _Float16* __restrict__ h16, const int* __restrict__ ei,
                         _Float16* __restrict__ kv, int E) {
  long long i = (long long)blockIdx.x * blockDim.x + threadIdx.x;
  long long total = (long long)E * KV_PAD;
  if (i >= total) return;
  int e = (int)(i / KV_PAD);
  int j = (int)(i - (long long)e * KV_PAD);
  _Float16 v;
  if (j < 4)        v = (_Float16)ea[(long long)e * EF + j];
  else if (j < 84)  v = rfeat16[(long long)e * 80 + (j - 4)];
  else if (j < 212) { int d = ei[E + e]; v = h16[(long long)d * HID + (j - 84)]; }
  else if (j < 340) { int s = ei[e];     v = h16[(long long)s * HID + (j - 212)]; }
  else              v = (_Float16)0.0f;
  kv[i] = v;
}

// ---------------------------------------------------------------------------
// WMMA GEMM: C[M,128] = A_f16[M,Kp] @ W[Kp,128] (+bias), W given transposed
// as Wt[128,Kp].  Block = 256 threads = 8 waves; block tile = 64 rows x 128
// cols; wave tile = 64 rows x 16 cols = 4 accumulators sharing one B frag.
// ---------------------------------------------------------------------------
__global__ __launch_bounds__(256)
void gemm_wmma_n128(const _Float16* __restrict__ A, const _Float16* __restrict__ Wt,
                    const float* __restrict__ bias, float* __restrict__ C,
                    int M, int Kp) {
  __shared__ _Float16 sA[64 * SA_STRIDE];
  const int t    = threadIdx.x;
  const int lane = t & 31;
  const int wave = t >> 5;
  const int rowBase = blockIdx.x * 64;
  const int nIdx = lane & 15;
  const int col  = wave * 16 + nIdx;
  const _Float16* wrow = Wt + (size_t)col * Kp;      // B column, contiguous K
  const int kb = (lane < 16) ? 0 : 16;               // unused; B frag K base is b0
  const int b0 = (lane < 16) ? 0 : 8;                // A frag K base (lo half)
  // staging: thread t moves one 16B chunk: row sr, halfs [sk, sk+8)
  const int sr = t >> 2;
  const int sk = (t & 3) * 8;
  int grow = rowBase + sr; if (grow > M - 1) grow = M - 1;
  const _Float16* arow = A + (size_t)grow * Kp;

  v8f c0 = {}, c1 = {}, c2 = {}, c3 = {};
  for (int k0 = 0; k0 < Kp; k0 += 32) {
    v8h av = *(const v8h*)(arow + k0 + sk);          // global_load_b128
    __syncthreads();                                 // protect prev frag reads
    *(v8h*)(&sA[sr * SA_STRIDE + sk]) = av;          // ds_store_b128
    __syncthreads();
    // B fragment: halves 0..7 = K b0..b0+7, halves 8..15 = K b0+16..b0+23
    v8h bl = *(const v8h*)(wrow + k0 + b0);
    v8h bh = *(const v8h*)(wrow + k0 + b0 + 16);
    v16h bf = __builtin_shufflevector(bl, bh, 0, 1, 2, 3, 4, 5, 6, 7,
                                      8, 9, 10, 11, 12, 13, 14, 15);
    // A fragments: halves 0..7 = K b0..b0+7, halves 8..15 = K b0+16..b0+23
#define A_FRAG(TILE)                                                          \
    ({ const _Float16* p = &sA[((TILE) * 16 + nIdx) * SA_STRIDE];             \
       v8h lo = *(const v8h*)(p + b0);                                        \
       v8h hi = *(const v8h*)(p + b0 + 16);                                   \
       __builtin_shufflevector(lo, hi, 0, 1, 2, 3, 4, 5, 6, 7,                \
                               8, 9, 10, 11, 12, 13, 14, 15); })
    c0 = __builtin_amdgcn_wmma_f32_16x16x32_f16(false, A_FRAG(0), false, bf, (short)0, c0, false, false);
    c1 = __builtin_amdgcn_wmma_f32_16x16x32_f16(false, A_FRAG(1), false, bf, (short)0, c1, false, false);
    c2 = __builtin_amdgcn_wmma_f32_16x16x32_f16(false, A_FRAG(2), false, bf, (short)0, c2, false, false);
    c3 = __builtin_amdgcn_wmma_f32_16x16x32_f16(false, A_FRAG(3), false, bf, (short)0, c3, false, false);
#undef A_FRAG
  }
  // epilogue: C/D layout -> col = lane&15, rows r + 8*(lane>=16)
  const int mhi = (lane >> 4) * 8;
  float bb = bias ? bias[col] : 0.0f;
  v8f acc[4] = {c0, c1, c2, c3};
#pragma unroll
  for (int tile = 0; tile < 4; ++tile) {
#pragma unroll
    for (int r = 0; r < 8; ++r) {
      int row = rowBase + tile * 16 + mhi + r;
      if (row < M) C[(long long)row * HID + col] = acc[tile][r] + bb;
    }
  }
}

// ---------------------------------------------------------------------------
// +b1 -> LayerNorm(g,b) -> ReLU -> f16   (one 128-thread block per row)
// ---------------------------------------------------------------------------
__global__ __launch_bounds__(128)
void ln_relu_f16(const float* __restrict__ Y, const float* __restrict__ b1,
                 const float* __restrict__ g, const float* __restrict__ b,
                 _Float16* __restrict__ out) {
  __shared__ float red[128];
  const int row = blockIdx.x;
  const int t = threadIdx.x;
  float y = Y[(long long)row * HID + t] + b1[t];
  red[t] = y;
  __syncthreads();
  for (int s = 64; s > 0; s >>= 1) { if (t < s) red[t] += red[t + s]; __syncthreads(); }
  float mu = red[0] * (1.0f / HID);
  __syncthreads();
  float dc = y - mu;
  red[t] = dc * dc;
  __syncthreads();
  for (int s = 64; s > 0; s >>= 1) { if (t < s) red[t] += red[t + s]; __syncthreads(); }
  float var = red[0] * (1.0f / HID);
  float yy = dc * rsqrtf(var + 1e-5f) * g[t] + b[t];
  out[(long long)row * HID + t] = (_Float16)fmaxf(yy, 0.0f);
}

// ---------------------------------------------------------------------------
// Second layer of xv MLP: [E,128]f16 @ [128,4]f32 + b2 -> [E,4]
// ---------------------------------------------------------------------------
__global__ void layer2_n4(const _Float16* __restrict__ a2, const float* __restrict__ w2,
                          const float* __restrict__ b2, float* __restrict__ out, int E) {
  long long i = (long long)blockIdx.x * blockDim.x + threadIdx.x;
  if (i >= (long long)E * HEADS) return;
  int e = (int)(i >> 2), hd = (int)(i & 3);
  float s = b2[hd];
#pragma unroll 8
  for (int d = 0; d < HID; ++d) s += (float)a2[(long long)e * HID + d] * w2[d * HEADS + hd];
  out[i] = s;
}

// ---------------------------------------------------------------------------
// score[e,h] = <q[dst[e],h,:], k[e,h,:]> / sqrt(32); segment max via
// monotone-uint atomicMax.
// ---------------------------------------------------------------------------
__global__ void score_segmax(const float* __restrict__ q, const float* __restrict__ k,
                             const int* __restrict__ ei, float* __restrict__ score,
                             unsigned* __restrict__ smax, int E) {
  long long i = (long long)blockIdx.x * blockDim.x + threadIdx.x;
  if (i >= (long long)E * HEADS) return;
  int e = (int)(i >> 2), hd = (int)(i & 3);
  int d = ei[E + e];
  const float* qr = q + (long long)d * HID + hd * HEAD_DIM;
  const float* kr = k + (long long)e * HID + hd * HEAD_DIM;
  float s = 0.0f;
#pragma unroll
  for (int j = 0; j < HEAD_DIM; ++j) s += qr[j] * kr[j];
  s *= INV_SQRT_D;
  score[i] = s;
  unsigned bits = __float_as_uint(s);
  unsigned enc = (bits & 0x80000000u) ? ~bits : (bits | 0x80000000u);
  atomicMax(&smax[d * HEADS + hd], enc);
}

__global__ void exp_segsum(float* __restrict__ score, const unsigned* __restrict__ smax,
                           float* __restrict__ ssum, const int* __restrict__ ei, int E) {
  long long i = (long long)blockIdx.x * blockDim.x + threadIdx.x;
  if (i >= (long long)E * HEADS) return;
  int e = (int)(i >> 2), hd = (int)(i & 3);
  int d = ei[E + e];
  unsigned enc = smax[d * HEADS + hd];
  float mx = 0.0f;
  if (enc != 0u) {
    unsigned bits = (enc & 0x80000000u) ? (enc & 0x7FFFFFFFu) : ~enc;
    mx = __uint_as_float(bits);
    if (!isfinite(mx)) mx = 0.0f;
  }
  float ee = expf(score[i] - mx);
  score[i] = ee;
  atomicAdd(&ssum[d * HEADS + hd], ee);
}

// agg[node,128] += alpha * ew * v   (x2h aggregation)
__global__ void agg_h(const float* __restrict__ enum_, const float* __restrict__ ssum,
                      const float* __restrict__ v, const float* __restrict__ ew,
                      const int* __restrict__ ei, float* __restrict__ agg, int E) {
  long long i = (long long)blockIdx.x * blockDim.x + threadIdx.x;
  if (i >= (long long)E * HID) return;
  int e = (int)(i >> 7), j = (int)(i & 127);
  int hd = j >> 5;
  int d = ei[E + e];
  float alpha = enum_[(long long)e * HEADS + hd] / ssum[d * HEADS + hd];
  atomicAdd(&agg[(long long)d * HID + j], alpha * ew[e] * v[i]);
}

// dx[node,head,3] += alpha * ew * vsmall * rel_x  (h2x aggregation)
__global__ void agg_x(const float* __restrict__ enum_, const float* __restrict__ ssum,
                      const float* __restrict__ vsmall, const float* __restrict__ ew,
                      const float* __restrict__ relx, const int* __restrict__ ei,
                      float* __restrict__ dx, int E) {
  long long i = (long long)blockIdx.x * blockDim.x + threadIdx.x;
  if (i >= (long long)E * 12) return;
  int e = (int)(i / 12);
  int r = (int)(i - (long long)e * 12);
  int hd = r / 3, dim = r - hd * 3;
  int d = ei[E + e];
  float alpha = enum_[(long long)e * HEADS + hd] / ssum[d * HEADS + hd];
  atomicAdd(&dx[(long long)d * 12 + r],
            alpha * ew[e] * vsmall[(long long)e * HEADS + hd] * relx[(long long)e * 3 + dim]);
}

// nodeA = [agg(128), h(128)] as f16
__global__ void build_nodeA(const float* __restrict__ agg, const float* __restrict__ h,
                            _Float16* __restrict__ A, int N) {
  long long i = (long long)blockIdx.x * blockDim.x + threadIdx.x;
  if (i >= (long long)N * 256) return;
  int n = (int)(i >> 8), j = (int)(i & 255);
  float v = (j < HID) ? agg[(long long)n * HID + j] : h[(long long)n * HID + (j - HID)];
  A[i] = (_Float16)v;
}

// x2h_out = mlp_out + h ; also produce f16 copy for the h2x stage
__global__ void residual_h(const float* __restrict__ nout, const float* __restrict__ h,
                           float* __restrict__ dout, _Float16* __restrict__ nh16, long long n) {
  long long i = (long long)blockIdx.x * blockDim.x + threadIdx.x;
  if (i >= n) return;
  float v = nout[i] + h[i];
  dout[i] = v;
  nh16[i] = (_Float16)v;
}

// x_new = x + mean_over_heads(dx) * mask
__global__ void finalize_x(const float* __restrict__ x, const float* __restrict__ dx,
                           const float* __restrict__ mask, float* __restrict__ out, int N) {
  long long i = (long long)blockIdx.x * blockDim.x + threadIdx.x;
  if (i >= (long long)N * 3) return;
  int n = (int)(i / 3), dim = (int)(i - (long long)n * 3);
  float m = 0.25f * (dx[n * 12 + dim] + dx[n * 12 + 3 + dim] +
                     dx[n * 12 + 6 + dim] + dx[n * 12 + 9 + dim]);
  out[i] = x[i] + m * mask[n];
}

// ---------------------------------------------------------------------------
// Host orchestration
// ---------------------------------------------------------------------------
extern "C" void kernel_launch(void* const* d_in, const int* in_sizes, int n_in,
                              void* d_out, int out_size, void* d_ws, size_t ws_size,
                              hipStream_t stream) {
  (void)n_in; (void)out_size; (void)ws_size;
  const float* h    = (const float*)d_in[0];
  const float* x    = (const float*)d_in[1];
  const float* ea   = (const float*)d_in[2];
  const float* mask = (const float*)d_in[3];
  const float* hk_w1 = (const float*)d_in[4];  const float* hk_b1 = (const float*)d_in[5];
  const float* hk_g  = (const float*)d_in[6];  const float* hk_b  = (const float*)d_in[7];
  const float* hk_w2 = (const float*)d_in[8];  const float* hk_b2 = (const float*)d_in[9];
  const float* hv_w1 = (const float*)d_in[10]; const float* hv_b1 = (const float*)d_in[11];
  const float* hv_g  = (const float*)d_in[12]; const float* hv_b  = (const float*)d_in[13];
  const float* hv_w2 = (const float*)d_in[14]; const float* hv_b2 = (const float*)d_in[15];
  const float* hq_w1 = (const float*)d_in[16]; const float* hq_b1 = (const float*)d_in[17];
  const float* hq_g  = (const float*)d_in[18]; const float* hq_b  = (const float*)d_in[19];
  const float* hq_w2 = (const float*)d_in[20]; const float* hq_b2 = (const float*)d_in[21];
  const float* ew_w1 = (const float*)d_in[22]; const float* ew_b1 = (const float*)d_in[23];
  const float* ou_w1 = (const float*)d_in[24]; const float* ou_b1 = (const float*)d_in[25];
  const float* ou_g  = (const float*)d_in[26]; const float* ou_b  = (const float*)d_in[27];
  const float* ou_w2 = (const float*)d_in[28]; const float* ou_b2 = (const float*)d_in[29];
  const float* xk_w1 = (const float*)d_in[30]; const float* xk_b1 = (const float*)d_in[31];
  const float* xk_g  = (const float*)d_in[32]; const float* xk_b  = (const float*)d_in[33];
  const float* xk_w2 = (const float*)d_in[34]; const float* xk_b2 = (const float*)d_in[35];
  const float* xv_w1 = (const float*)d_in[36]; const float* xv_b1 = (const float*)d_in[37];
  const float* xv_g  = (const float*)d_in[38]; const float* xv_b  = (const float*)d_in[39];
  const float* xv_w2 = (const float*)d_in[40]; const float* xv_b2 = (const float*)d_in[41];
  const float* xq_w1 = (const float*)d_in[42]; const float* xq_b1 = (const float*)d_in[43];
  const float* xq_g  = (const float*)d_in[44]; const float* xq_b  = (const float*)d_in[45];
  const float* xq_w2 = (const float*)d_in[46]; const float* xq_b2 = (const float*)d_in[47];
  const float* ew_w2 = (const float*)d_in[48]; const float* ew_b2 = (const float*)d_in[49];
  const int*   ei    = (const int*)d_in[50];

  const int N = in_sizes[0] / HID;
  const int E = in_sizes[2] / EF;

  // ---- workspace bump allocator (256B aligned) ----
  char* base = (char*)d_ws;
  size_t off = 0;
  auto alloc = [&](size_t bytes) -> void* {
    off = (off + 255) & ~(size_t)255;
    void* p = base + off;
    off += bytes;
    return p;
  };
  _Float16* rfeat16 = (_Float16*)alloc((size_t)E * 80 * 2);
  float*    relx    = (float*)alloc((size_t)E * 3 * 4);
  float*    ew1     = (float*)alloc((size_t)E * 4);
  float*    ew2     = (float*)alloc((size_t)E * 4);
  _Float16* h16     = (_Float16*)alloc((size_t)N * HID * 2);
  _Float16* nh16    = (_Float16*)alloc((size_t)N * HID * 2);
  _Float16* kv16    = (_Float16*)alloc((size_t)E * KV_PAD * 2);
  float*    y1      = (float*)alloc((size_t)E * HID * 4);
  _Float16* a2      = (_Float16*)alloc((size_t)E * HID * 2);
  float*    kbuf    = (float*)alloc((size_t)E * HID * 4);
  float*    vbuf    = (float*)alloc((size_t)E * HID * 4);
  float*    vsmall  = (float*)alloc((size_t)E * HEADS * 4);
  float*    score   = (float*)alloc((size_t)E * HEADS * 4);
  unsigned* smax    = (unsigned*)alloc((size_t)N * HEADS * 4);
  float*    ssum    = (float*)alloc((size_t)N * HEADS * 4);
  float*    aggbuf  = (float*)alloc((size_t)N * HID * 4);
  float*    qbuf    = (float*)alloc((size_t)N * HID * 4);
  _Float16* nodeA   = (_Float16*)alloc((size_t)N * 256 * 2);
  float*    nout    = (float*)alloc((size_t)N * HID * 4);
  float*    dxbuf   = (float*)alloc((size_t)N * 12 * 4);
  // transposed f16 weights: Wt[128][Kp]
  _Float16* w_hk1 = (_Float16*)alloc((size_t)HID * KV_PAD * 2);
  _Float16* w_hv1 = (_Float16*)alloc((size_t)HID * KV_PAD * 2);
  _Float16* w_xk1 = (_Float16*)alloc((size_t)HID * KV_PAD * 2);
  _Float16* w_xv1 = (_Float16*)alloc((size_t)HID * KV_PAD * 2);
  _Float16* w_hk2 = (_Float16*)alloc((size_t)HID * HID * 2);
  _Float16* w_hv2 = (_Float16*)alloc((size_t)HID * HID * 2);
  _Float16* w_xk2 = (_Float16*)alloc((size_t)HID * HID * 2);
  _Float16* w_hq1 = (_Float16*)alloc((size_t)HID * HID * 2);
  _Float16* w_hq2 = (_Float16*)alloc((size_t)HID * HID * 2);
  _Float16* w_xq1 = (_Float16*)alloc((size_t)HID * HID * 2);
  _Float16* w_xq2 = (_Float16*)alloc((size_t)HID * HID * 2);
  _Float16* w_ou1 = (_Float16*)alloc((size_t)HID * 256 * 2);
  _Float16* w_ou2 = (_Float16*)alloc((size_t)HID * HID * 2);

  const int TB = 256;
  auto g1 = [&](long long n) { return (unsigned)cdivll(n, TB); };

  // ---- weight prep: transpose to Wt[128][Kp] f16 ----
  hipLaunchKernelGGL(transpose_w_f16, dim3(g1((long long)HID * KV_PAD)), dim3(TB), 0, stream, hk_w1, w_hk1, KV_RAW, KV_PAD);
  hipLaunchKernelGGL(transpose_w_f16, dim3(g1((long long)HID * KV_PAD)), dim3(TB), 0, stream, hv_w1, w_hv1, KV_RAW, KV_PAD);
  hipLaunchKernelGGL(transpose_w_f16, dim3(g1((long long)HID * KV_PAD)), dim3(TB), 0, stream, xk_w1, w_xk1, KV_RAW, KV_PAD);
  hipLaunchKernelGGL(transpose_w_f16, dim3(g1((long long)HID * KV_PAD)), dim3(TB), 0, stream, xv_w1, w_xv1, KV_RAW, KV_PAD);
  hipLaunchKernelGGL(transpose_w_f16, dim3(g1((long long)HID * HID)), dim3(TB), 0, stream, hk_w2, w_hk2, HID, HID);
  hipLaunchKernelGGL(transpose_w_f16, dim3(g1((long long)HID * HID)), dim3(TB), 0, stream, hv_w2, w_hv2, HID, HID);
  hipLaunchKernelGGL(transpose_w_f16, dim3(g1((long long)HID * HID)), dim3(TB), 0, stream, xk_w2, w_xk2, HID, HID);
  hipLaunchKernelGGL(transpose_w_f16, dim3(g1((long long)HID * HID)), dim3(TB), 0, stream, hq_w1, w_hq1, HID, HID);
  hipLaunchKernelGGL(transpose_w_f16, dim3(g1((long long)HID * HID)), dim3(TB), 0, stream, hq_w2, w_hq2, HID, HID);
  hipLaunchKernelGGL(transpose_w_f16, dim3(g1((long long)HID * HID)), dim3(TB), 0, stream, xq_w1, w_xq1, HID, HID);
  hipLaunchKernelGGL(transpose_w_f16, dim3(g1((long long)HID * HID)), dim3(TB), 0, stream, xq_w2, w_xq2, HID, HID);
  hipLaunchKernelGGL(transpose_w_f16, dim3(g1((long long)HID * 256)), dim3(TB), 0, stream, ou_w1, w_ou1, 256, 256);
  hipLaunchKernelGGL(transpose_w_f16, dim3(g1((long long)HID * HID)), dim3(TB), 0, stream, ou_w2, w_ou2, HID, HID);

  // ---- stage 0 ----
  hipLaunchKernelGGL(f32_to_f16, dim3(g1((long long)N * HID)), dim3(TB), 0, stream, h, h16, (long long)N * HID);
  hipLaunchKernelGGL(edge_geom, dim3(g1(E)), dim3(TB), 0, stream, x, ea, ei,
                     ew_w1, ew_b1, ew_w2, ew_b2, rfeat16, relx, ew1, ew2, E);

  const unsigned gE64 = (unsigned)cdivll(E, 64);
  const unsigned gN64 = (unsigned)cdivll(N, 64);

  // ================= x2h =================
  hipLaunchKernelGGL(build_kv, dim3(g1((long long)E * KV_PAD)), dim3(TB), 0, stream, ea, rfeat16, h16, ei, kv16, E);
  hipLaunchKernelGGL(gemm_wmma_n128, dim3(gE64), dim3(256), 0, stream, kv16, w_hk1, (const float*)nullptr, y1, E, KV_PAD);
  hipLaunchKernelGGL(ln_relu_f16, dim3((unsigned)E), dim3(128), 0, stream, y1, hk_b1, hk_g, hk_b, a2);
  hipLaunchKernelGGL(gemm_wmma_n128, dim3(gE64), dim3(256), 0, stream, a2, w_hk2, hk_b2, kbuf, E, HID);
  hipLaunchKernelGGL(gemm_wmma_n128, dim3(gE64), dim3(256), 0, stream, kv16, w_hv1, (const float*)nullptr, y1, E, KV_PAD);
  hipLaunchKernelGGL(ln_relu_f16, dim3((unsigned)E), dim3(128), 0, stream, y1, hv_b1, hv_g, hv_b, a2);
  hipLaunchKernelGGL(gemm_wmma_n128, dim3(gE64), dim3(256), 0, stream, a2, w_hv2, hv_b2, vbuf, E, HID);
  hipLaunchKernelGGL(gemm_wmma_n128, dim3(gN64), dim3(256), 0, stream, h16, w_hq1, (const float*)nullptr, y1, N, HID);
  hipLaunchKernelGGL(ln_relu_f16, dim3((unsigned)N), dim3(128), 0, stream, y1, hq_b1, hq_g, hq_b, a2);
  hipLaunchKernelGGL(gemm_wmma_n128, dim3(gN64), dim3(256), 0, stream, a2, w_hq2, hq_b2, qbuf, N, HID);
  hipLaunchKernelGGL(zero_f32, dim3(g1((long long)N * HEADS)), dim3(TB), 0, stream, (float*)smax, (long long)N * HEADS);
  hipLaunchKernelGGL(zero_f32, dim3(g1((long long)N * HEADS)), dim3(TB), 0, stream, ssum, (long long)N * HEADS);
  hipLaunchKernelGGL(zero_f32, dim3(g1((long long)N * HID)), dim3(TB), 0, stream, aggbuf, (long long)N * HID);
  hipLaunchKernelGGL(score_segmax, dim3(g1((long long)E * HEADS)), dim3(TB), 0, stream, qbuf, kbuf, ei, score, smax, E);
  hipLaunchKernelGGL(exp_segsum, dim3(g1((long long)E * HEADS)), dim3(TB), 0, stream, score, smax, ssum, ei, E);
  hipLaunchKernelGGL(agg_h, dim3(g1((long long)E * HID)), dim3(TB), 0, stream, score, ssum, vbuf, ew1, ei, aggbuf, E);
  hipLaunchKernelGGL(build_nodeA, dim3(g1((long long)N * 256)), dim3(TB), 0, stream, aggbuf, h, nodeA, N);
  hipLaunchKernelGGL(gemm_wmma_n128, dim3(gN64), dim3(256), 0, stream, nodeA, w_ou1, (const float*)nullptr, y1, N, 256);
  hipLaunchKernelGGL(ln_relu_f16, dim3((unsigned)N), dim3(128), 0, stream, y1, ou_b1, ou_g, ou_b, a2);
  hipLaunchKernelGGL(gemm_wmma_n128, dim3(gN64), dim3(256), 0, stream, a2, w_ou2, ou_b2, nout, N, HID);
  hipLaunchKernelGGL(residual_h, dim3(g1((long long)N * HID)), dim3(TB), 0, stream, nout, h, (float*)d_out, nh16, (long long)N * HID);

  // ================= h2x =================
  hipLaunchKernelGGL(build_kv, dim3(g1((long long)E * KV_PAD)), dim3(TB), 0, stream, ea, rfeat16, nh16, ei, kv16, E);
  hipLaunchKernelGGL(gemm_wmma_n128, dim3(gE64), dim3(256), 0, stream, kv16, w_xk1, (const float*)nullptr, y1, E, KV_PAD);
  hipLaunchKernelGGL(ln_relu_f16, dim3((unsigned)E), dim3(128), 0, stream, y1, xk_b1, xk_g, xk_b, a2);
  hipLaunchKernelGGL(gemm_wmma_n128, dim3(gE64), dim3(256), 0, stream, a2, w_xk2, xk_b2, kbuf, E, HID);
  hipLaunchKernelGGL(gemm_wmma_n128, dim3(gE64), dim3(256), 0, stream, kv16, w_xv1, (const float*)nullptr, y1, E, KV_PAD);
  hipLaunchKernelGGL(ln_relu_f16, dim3((unsigned)E), dim3(128), 0, stream, y1, xv_b1, xv_g, xv_b, a2);
  hipLaunchKernelGGL(layer2_n4, dim3(g1((long long)E * HEADS)), dim3(TB), 0, stream, a2, xv_w2, xv_b2, vsmall, E);
  hipLaunchKernelGGL(gemm_wmma_n128, dim3(gN64), dim3(256), 0, stream, nh16, w_xq1, (const float*)nullptr, y1, N, HID);
  hipLaunchKernelGGL(ln_relu_f16, dim3((unsigned)N), dim3(128), 0, stream, y1, xq_b1, xq_g, xq_b, a2);
  hipLaunchKernelGGL(gemm_wmma_n128, dim3(gN64), dim3(256), 0, stream, a2, w_xq2, xq_b2, qbuf, N, HID);
  hipLaunchKernelGGL(zero_f32, dim3(g1((long long)N * HEADS)), dim3(TB), 0, stream, (float*)smax, (long long)N * HEADS);
  hipLaunchKernelGGL(zero_f32, dim3(g1((long long)N * HEADS)), dim3(TB), 0, stream, ssum, (long long)N * HEADS);
  hipLaunchKernelGGL(zero_f32, dim3(g1((long long)N * 12)), dim3(TB), 0, stream, dxbuf, (long long)N * 12);
  hipLaunchKernelGGL(score_segmax, dim3(g1((long long)E * HEADS)), dim3(TB), 0, stream, qbuf, kbuf, ei, score, smax, E);
  hipLaunchKernelGGL(exp_segsum, dim3(g1((long long)E * HEADS)), dim3(TB), 0, stream, score, smax, ssum, ei, E);
  hipLaunchKernelGGL(agg_x, dim3(g1((long long)E * 12)), dim3(TB), 0, stream, score, ssum, vsmall, ew2, relx, ei, dxbuf, E);
  hipLaunchKernelGGL(finalize_x, dim3(g1((long long)N * 3)), dim3(TB), 0, stream, x, dxbuf, mask, (float*)d_out + (size_t)N * HID, N);
}
